// PhysicsEngine_41351945126384
// MI455X (gfx1250) — matrix operand order repaired and checked by reading, not compiled, not run on hardware
//
// MI455X / gfx1250 implementation of the PhysicsEngine reference.
//
// Roofline: dominant cost = edge-wise MLPs over E=262144 edges (K=384->128->128->128,
// ~86 GFLOP) + encoders/GNO MLPs; at 23.3 TB/s this is gather/bandwidth bound, so:
//   * every row-major X@W MLP runs through v_wmma_f32_16x16x32_bf16 (f32 accum),
//   * concat(nf[dst], nf[src], ef) gathers are fused into the WMMA A loader
//     (the [E,384] matrix is never materialized),
//   * weights are staged in LDS pre-swizzled into WMMA B-fragment order so each
//     fragment is one 32B v16bf read (2x ds_load_b128); staging itself is
//     float4-vectorized (2x global_load_b128 + 4x ds_store_b32 per 8 weights),
//   * A K-pairs load as float2 (global_load_b64) on the aligned hot paths,
//   * kernel is templated on NT tiles so the WMMA loop has no per-tile branches,
//   * all GEMM index math is 32-bit (max offset ~1e8 << 2^31),
//   * segment sums use global_atomic_add_f32; the tiny 32x32 FNO spectral path
//     uses direct 32-pt DFT kernels; all 1x1 convs reuse the WMMA GEMM.
//
// Param d_in ordering assumes jax tree-flatten (sorted dict keys); see enum.

#include <hip/hip_runtime.h>
#include <math.h>

typedef __attribute__((ext_vector_type(16))) __bf16 v16bf;
typedef __attribute__((ext_vector_type(8)))  float  v8f;

#define GRID_G 32
#define KXN 17
#define PI_F 3.14159265358979f

static inline int cdiv(long long a, long long b) { return (int)((a + b - 1) / b); }

__device__ __forceinline__ float act_apply(float x, int act) {
  if (act == 1) return fmaxf(x, 0.f);
  if (act == 2) {  // tanh-approx gelu (jax.nn.gelu default)
    float u = 0.7978845608028654f * (x + 0.044715f * x * x * x);
    return 0.5f * x * (1.f + tanhf(u));
  }
  if (act == 3) return tanhf(x);
  return x;
}

__device__ __forceinline__ unsigned int pack_bf16(float a, float b) {
  __bf16 ha = (__bf16)a, hb = (__bf16)b;
  unsigned short ua = __builtin_bit_cast(unsigned short, ha);
  unsigned short ub = __builtin_bit_cast(unsigned short, hb);
  return (unsigned int)ua | ((unsigned int)ub << 16);
}

// ---------------------------------------------------------------------------
// Generic WMMA GEMM: C[M,Nout] = act(A[M,K] @ W[K,Nout] + bias) + residual
// A is either plain (A,lda) or a fused concat-gather of up to 3 segments:
//   k<c0        : A0[(i0?i0[row]:row)*c0 + k]
//   k<c0+c1     : A1[(i1?i1[row]:row)*c1 + k-c0]
//   else        : A2[row*c2 + k-c0-c1]
// vec==1 -> all strides even & bases 8B aligned: K-pairs load as float2.
// ---------------------------------------------------------------------------
struct GemmArgs {
  const float* A; int lda;
  const float* A0; const int* i0; int c0;
  const float* A1; const int* i1; int c1;
  const float* A2; int c2;
  const float* W; const float* bias; const float* residual;
  float* C; int ldc;
  int M, K, Nout, act, vec, c01;
};

template <int NT>
__global__ __launch_bounds__(256) void gemm_wmma_kernel(GemmArgs g) {
  // Weights staged pre-swizzled into B-fragment order:
  // frag f = ktile*NT + ntile; per (f, lane) the 16 bf16 a lane needs are
  // contiguous & 32B aligned -> one v16bf LDS read per fragment.
  __shared__ __align__(32) unsigned int sB[4 * NT * 32 * 8];  // NT=8 -> 32KB
  const int lane = threadIdx.x & 31;
  const int wave = threadIdx.x >> 5;
  const int r    = lane & 15;
  const int half = lane >> 4;
  const int mbase = (blockIdx.x * 8 + wave) * 16;
  const int rowM  = mbase + r;

  // Hoist gather/segment resolution out of the K loop (32-bit offsets).
  const float* s0 = nullptr;
  const float* s1 = nullptr;
  const float* s2 = nullptr;
  if (rowM < g.M) {
    if (g.A) {
      s0 = g.A + rowM * g.lda;
    } else {
      int r0 = g.i0 ? g.i0[rowM] : rowM;
      s0 = g.A0 + r0 * g.c0;
      if (g.A1) { int r1 = g.i1 ? g.i1[rowM] : rowM; s1 = g.A1 + r1 * g.c1; }
      if (g.A2) s2 = g.A2 + rowM * g.c2;
    }
  }

  auto loadA2 = [&](int k) -> float2 {
    float2 z; z.x = 0.f; z.y = 0.f;
    if (rowM >= g.M || k >= g.K) return z;
    const float* p;
    if (g.A)            p = s0 + k;
    else if (k < g.c0)  p = s0 + k;
    else if (k < g.c01) p = s1 + (k - g.c0);
    else                p = s2 + (k - g.c01);
    if (g.vec) return *(const float2*)p;  // stride/base even -> 8B aligned
    float2 out; out.x = p[0]; out.y = (k + 1 < g.K) ? p[1] : 0.f;
    return out;
  };

  v8f acc[NT] = {};
  const bool quad = (g.Nout & 3) == 0;  // Nout%4==0 -> Nout == NT*16 exactly

  for (int kc = 0; kc < g.K; kc += 128) {
    __syncthreads();
    if (quad) {
      // Vectorized staging: two adjacent W rows as float4 (coalesced along n,
      // 16B aligned), pack K-pairs, 4x ds_store_b32 into fragment layout.
      const int ng = g.Nout >> 2;        // float4 groups per row
      const int units = 64 * ng;         // 64 K-pairs per 128-chunk
      for (int u = threadIdx.x; u < units; u += 256) {
        int q = u / ng, n = (u % ng) << 2;
        int kk = kc + 2 * q;
        float4 a4; a4.x = a4.y = a4.z = a4.w = 0.f;
        float4 b4; b4.x = b4.y = b4.z = b4.w = 0.f;
        if (kk < g.K)     a4 = *(const float4*)(g.W + kk * g.Nout + n);
        if (kk + 1 < g.K) b4 = *(const float4*)(g.W + (kk + 1) * g.Nout + n);
        int ktile = q >> 4, kh = (q >> 3) & 1, v = q & 7;
        int jb = (((ktile * NT + (n >> 4)) * 32) + kh * 16 + (n & 15)) * 8 + v;
        sB[jb]      = pack_bf16(a4.x, b4.x);
        sB[jb + 8]  = pack_bf16(a4.y, b4.y);
        sB[jb + 16] = pack_bf16(a4.z, b4.z);
        sB[jb + 24] = pack_bf16(a4.w, b4.w);
      }
    } else {
      // Scalar fallback for Nout % 4 != 0 (only the tiny Nout=3/2 layers).
      for (int j = threadIdx.x; j < 4 * NT * 32 * 8; j += 256) {
        int v  = j & 7;
        int l  = (j >> 3) & 31;
        int f  = j >> 8;
        int ktile = f / NT, ntile = f % NT;
        int kk = kc + ktile * 32 + 2 * v + 16 * (l >> 4);
        int n  = ntile * 16 + (l & 15);
        float w0 = (kk     < g.K && n < g.Nout) ? g.W[kk * g.Nout + n] : 0.f;
        float w1 = (kk + 1 < g.K && n < g.Nout) ? g.W[(kk + 1) * g.Nout + n] : 0.f;
        sB[j] = pack_bf16(w0, w1);
      }
    }
    __syncthreads();

    for (int kt = 0; kt < 128 && kc + kt < g.K; kt += 32) {
      const int ktile = kt >> 5;
      // A fragment, contiguous k order (k = 16*half + s, matches B staging), float2 per K-pair.
      v16bf a;
#pragma unroll
      for (int v = 0; v < 8; ++v) {
        int k0 = kc + kt + 2 * v + 16 * half;
        float2 x = loadA2(k0);
        a[2 * v]     = (__bf16)x.x;
        a[2 * v + 1] = (__bf16)x.y;
      }
#pragma unroll
      for (int t = 0; t < NT; ++t) {
        v16bf b = reinterpret_cast<const v16bf*>(sB)[(ktile * NT + t) * 32 + lane];
        acc[t] = __builtin_amdgcn_wmma_f32_16x16x32_bf16(
            false, a, false, b, (short)0, acc[t], false, false);
      }
    }
  }

#pragma unroll
  for (int t = 0; t < NT; ++t) {
#pragma unroll
    for (int v = 0; v < 8; ++v) {
      int row = mbase + v + 8 * half;
      int col = t * 16 + r;
      if (row < g.M && col < g.Nout) {
        float val = acc[t][v];
        if (g.bias) val += g.bias[col];
        val = act_apply(val, g.act);
        if (g.residual) val += g.residual[row * g.ldc + col];
        g.C[row * g.ldc + col] = val;
      }
    }
  }
}

// ---------------------------------------------------------------------------
// LayerNorm over C=128 features, wave-per-row, optional per-row scale.
// ---------------------------------------------------------------------------
__global__ void ln_kernel(float* X, const float* gam, const float* bet,
                          const float* rowscale, int M, int C) {
  int wave = threadIdx.x >> 5, lane = threadIdx.x & 31;
  int row = blockIdx.x * (blockDim.x >> 5) + wave;
  if (row >= M) return;
  float v[4];
  float s = 0.f;
#pragma unroll
  for (int j = 0; j < 4; ++j) {
    int c = lane + 32 * j;
    v[j] = (c < C) ? X[(size_t)row * C + c] : 0.f;
    s += v[j];
  }
#pragma unroll
  for (int off = 16; off >= 1; off >>= 1) s += __shfl_xor(s, off, 32);
  float mu = s / (float)C;
  float q = 0.f;
#pragma unroll
  for (int j = 0; j < 4; ++j) {
    int c = lane + 32 * j;
    if (c < C) { float d = v[j] - mu; q += d * d; }
  }
#pragma unroll
  for (int off = 16; off >= 1; off >>= 1) q += __shfl_xor(q, off, 32);
  float rstd = rsqrtf(q / (float)C + 1e-5f);
  float sc = rowscale ? rowscale[row] : 1.f;
#pragma unroll
  for (int j = 0; j < 4; ++j) {
    int c = lane + 32 * j;
    if (c < C) X[(size_t)row * C + c] = ((v[j] - mu) * rstd * gam[c] + bet[c]) * sc;
  }
}

// ---------------------------------------------------------------------------
// Scatter / elementwise helpers
// ---------------------------------------------------------------------------
__global__ void scatter_add_kernel(const float* src, const int* idx, float* dst,
                                   int E, int C) {
  long long t = (long long)blockIdx.x * blockDim.x + threadIdx.x;
  if (t >= (long long)E * C) return;
  int e = (int)(t / C), c = (int)(t % C);
  atomicAdd(&dst[(size_t)idx[e] * C + c], src[t]);
}

__global__ void ewise_act_kernel(float* dst, const float* src, int n, int act) {
  int i = blockIdx.x * blockDim.x + threadIdx.x;
  if (i < n) dst[i] = act_apply(src[i], act);
}

__global__ void add_inplace_kernel(float* dst, const float* src, long long n) {
  long long i = (long long)blockIdx.x * blockDim.x + threadIdx.x;
  if (i < n) dst[i] += src[i];
}

__global__ void div_mean_kernel(const float* sum, const float* cnt, float* out,
                                int n, int C) {
  long long t = (long long)blockIdx.x * blockDim.x + threadIdx.x;
  if (t >= (long long)n * C) return;
  int i = (int)(t / C);
  out[t] = sum[t] / fmaxf(cnt[i], 1.f);
}

__global__ void pos2d_kernel(const float* rp, const float* W, const float* b,
                             float* out, int n) {
  int i = blockIdx.x * blockDim.x + threadIdx.x;
  if (i >= n) return;
#pragma unroll
  for (int c = 0; c < 2; ++c) {
    float s = b[c];
#pragma unroll
    for (int d = 0; d < 3; ++d) s += rp[i * 3 + d] * W[d * 2 + c];
    out[i * 2 + c] = tanhf(s);
  }
}

__global__ void kin_build_kernel(const int* q, const int* p, const float* p2d,
                                 const float* lq, const float* out3, float* kin,
                                 int P) {
  int e = blockIdx.x * blockDim.x + threadIdx.x;
  if (e >= P) return;
  int qq = q[e], pp = p[e];
  kin[e * 5 + 0] = p2d[pp * 2 + 0] - lq[qq * 2 + 0];
  kin[e * 5 + 1] = p2d[pp * 2 + 1] - lq[qq * 2 + 1];
  kin[e * 5 + 2] = out3[pp * 3 + 0];
  kin[e * 5 + 3] = out3[pp * 3 + 1];
  kin[e * 5 + 4] = out3[pp * 3 + 2];
}

__global__ void kd_build_kernel(const int* q, const int* p, const float* p2d,
                                const float* lq, float* kd, int P) {
  int e = blockIdx.x * blockDim.x + threadIdx.x;
  if (e >= P) return;
  int qq = q[e], pp = p[e];  // p indexes latent queries, q indexes particles
  kd[e * 2 + 0] = lq[pp * 2 + 0] - p2d[qq * 2 + 0];
  kd[e * 2 + 1] = lq[pp * 2 + 1] - p2d[qq * 2 + 1];
}

__global__ void gno_in_accum_kernel(const float* kv, const int* q, float* sum,
                                    float* cnt, int P) {
  long long t = (long long)blockIdx.x * blockDim.x + threadIdx.x;
  if (t >= (long long)P * 3) return;
  int e = (int)(t / 3), c = (int)(t % 3);
  atomicAdd(&sum[(size_t)q[e] * 3 + c], kv[t]);
  if (c == 0) atomicAdd(&cnt[q[e]], 1.f);
}

__global__ void gno_out_accum_kernel(const float* kout, const int* q,
                                     const int* p, const float* latFlat,
                                     float* osum, float* ocnt, int P) {
  long long t = (long long)blockIdx.x * blockDim.x + threadIdx.x;
  if (t >= (long long)P * 128) return;
  int e = (int)(t >> 7), c = (int)(t & 127);
  float v = kout[t] * latFlat[(size_t)p[e] * 128 + c];
  atomicAdd(&osum[(size_t)q[e] * 128 + c], v);
  if (c == 0) atomicAdd(&ocnt[q[e]], 1.f);
}

__global__ void transpose_pm_cm_kernel(const float* pm, float* cm, int HW, int C) {
  long long t = (long long)blockIdx.x * blockDim.x + threadIdx.x;
  if (t >= (long long)HW * C) return;
  int s = (int)(t / C), c = (int)(t % C);
  cm[(size_t)c * HW + s] = pm[t];
}

// ---------------------------------------------------------------------------
// Direct 32-point DFT passes for the FNO spectral conv (32x32 grid, 32 ch)
// ---------------------------------------------------------------------------
__global__ void dft_row_fwd_kernel(const float* x, float* Yr, float* Yi, int C) {
  int t = blockIdx.x * blockDim.x + threadIdx.x;
  if (t >= C * GRID_G * KXN) return;
  int kx = t % KXN, h = (t / KXN) % GRID_G, c = t / (KXN * GRID_G);
  float sr = 0.f, si = 0.f;
  for (int w = 0; w < GRID_G; ++w) {
    float v = x[(h * GRID_G + w) * C + c];
    float ang = -2.f * PI_F * (float)(kx * w) / (float)GRID_G;
    float sn, cs; sincosf(ang, &sn, &cs);
    sr += v * cs; si += v * sn;
  }
  Yr[t] = sr; Yi[t] = si;
}

__global__ void dft_col_fwd_kernel(const float* Yr, const float* Yi, float* Xr,
                                   float* Xi, int C) {
  int t = blockIdx.x * blockDim.x + threadIdx.x;
  if (t >= C * GRID_G * KXN) return;
  int kx = t % KXN, ky = (t / KXN) % GRID_G, c = t / (KXN * GRID_G);
  float sr = 0.f, si = 0.f;
  for (int h = 0; h < GRID_G; ++h) {
    float yr = Yr[(c * GRID_G + h) * KXN + kx];
    float yi = Yi[(c * GRID_G + h) * KXN + kx];
    float ang = -2.f * PI_F * (float)(ky * h) / (float)GRID_G;
    float sn, cs; sincosf(ang, &sn, &cs);
    sr += yr * cs - yi * sn;
    si += yr * sn + yi * cs;
  }
  Xr[t] = sr; Xi[t] = si;
}

__global__ void spec_mul_kernel(const float* Xr, const float* Xi,
                                const float* w1r, const float* w1i,
                                const float* w2r, const float* w2i,
                                float* Or_, float* Oi_) {
  int t = blockIdx.x * blockDim.x + threadIdx.x;
  if (t >= 32 * GRID_G * KXN) return;
  int kx = t % KXN, ky = (t / KXN) % GRID_G, o = t / (KXN * GRID_G);
  if (kx >= 16) { Or_[t] = 0.f; Oi_[t] = 0.f; return; }
  const float* wr; const float* wi; int row;
  if (ky < 16) { wr = w1r; wi = w1i; row = ky; }
  else         { wr = w2r; wi = w2i; row = ky - 16; }
  float sr = 0.f, si = 0.f;
  for (int i = 0; i < 32; ++i) {
    float xr = Xr[(i * GRID_G + ky) * KXN + kx];
    float xi = Xi[(i * GRID_G + ky) * KXN + kx];
    float wwr = wr[((i * 32 + o) * 16 + row) * 16 + kx];
    float wwi = wi[((i * 32 + o) * 16 + row) * 16 + kx];
    sr += xr * wwr - xi * wwi;
    si += xr * wwi + xi * wwr;
  }
  Or_[t] = sr; Oi_[t] = si;
}

__global__ void dft_col_inv_kernel(const float* Or_, const float* Oi_, float* Zr,
                                   float* Zi) {
  int t = blockIdx.x * blockDim.x + threadIdx.x;
  if (t >= 32 * GRID_G * KXN) return;
  int kx = t % KXN, h = (t / KXN) % GRID_G, o = t / (KXN * GRID_G);
  float sr = 0.f, si = 0.f;
  for (int ky = 0; ky < GRID_G; ++ky) {
    float orr = Or_[(o * GRID_G + ky) * KXN + kx];
    float oii = Oi_[(o * GRID_G + ky) * KXN + kx];
    float ang = 2.f * PI_F * (float)(ky * h) / (float)GRID_G;
    float sn, cs; sincosf(ang, &sn, &cs);
    sr += orr * cs - oii * sn;
    si += orr * sn + oii * cs;
  }
  Zr[t] = sr; Zi[t] = si;
}

__global__ void dft_row_inv_kernel(const float* Zr, const float* Zi, float* out,
                                   int C) {
  int t = blockIdx.x * blockDim.x + threadIdx.x;
  if (t >= C * GRID_G * GRID_G) return;
  int w = t % GRID_G, h = (t / GRID_G) % GRID_G, o = t / (GRID_G * GRID_G);
  float s = 0.f;
  for (int kx = 0; kx < KXN; ++kx) {
    float wgt = (kx == 0 || kx == 16) ? 1.f : 2.f;  // hermitian symmetry
    float ang = 2.f * PI_F * (float)(kx * w) / (float)GRID_G;
    float sn, cs; sincosf(ang, &sn, &cs);
    s += wgt * (Zr[(o * GRID_G + h) * KXN + kx] * cs -
                Zi[(o * GRID_G + h) * KXN + kx] * sn);
  }
  out[(h * GRID_G + w) * C + o] = s * (1.f / (GRID_G * GRID_G));
}

// ---------------------------------------------------------------------------
// Host side
// ---------------------------------------------------------------------------
static void launch_gemm(hipStream_t st, GemmArgs& g) {
  g.c01 = g.c0 + g.c1;
  if (g.A) g.vec = (g.lda % 2 == 0 && g.K % 2 == 0) ? 1 : 0;
  else     g.vec = (g.c0 % 2 == 0 && g.c1 % 2 == 0 && g.c2 % 2 == 0) ? 1 : 0;
  int blocks = cdiv(g.M, 128);
  int nt = (g.Nout + 15) >> 4;
  if (nt <= 1)      gemm_wmma_kernel<1><<<blocks, 256, 0, st>>>(g);
  else if (nt <= 2) gemm_wmma_kernel<2><<<blocks, 256, 0, st>>>(g);
  else if (nt <= 4) gemm_wmma_kernel<4><<<blocks, 256, 0, st>>>(g);
  else              gemm_wmma_kernel<8><<<blocks, 256, 0, st>>>(g);
}

static void run_gemm(hipStream_t st, const float* A, int lda, const float* W,
                     const float* bias, const float* residual, float* C, int M,
                     int K, int Nout, int act) {
  GemmArgs g{};
  g.A = A; g.lda = lda; g.W = W; g.bias = bias; g.residual = residual;
  g.C = C; g.ldc = Nout; g.M = M; g.K = K; g.Nout = Nout; g.act = act;
  launch_gemm(st, g);
}

static void run_gemm_cat(hipStream_t st, const float* A0, const int* i0, int c0,
                         const float* A1, const int* i1, int c1, const float* A2,
                         int c2, const float* W, const float* bias, float* C,
                         int M, int Nout, int act) {
  GemmArgs g{};
  g.A0 = A0; g.i0 = i0; g.c0 = c0;
  g.A1 = A1; g.i1 = i1; g.c1 = c1;
  g.A2 = A2; g.c2 = c2;
  g.W = W; g.bias = bias; g.C = C; g.ldc = Nout;
  g.M = M; g.K = c0 + c1 + c2; g.Nout = Nout; g.act = act;
  launch_gemm(st, g);
}

// Param leaf order: jax tree-flatten, dict keys sorted, lists in order.
enum {
  P_W2D = 0, P_B2D,
  P_EIN_W0, P_EIN_W1, P_EIN_W2, P_EIN_B0, P_EIN_B1, P_EIN_B2, P_EIN_BE, P_EIN_G,
  P_EMBED,
  P_BLK0,                           // 2 blocks x {m1W,m1b,m2W,m2b,msW,msb,skW,skb,w1i,w1r,w2i,w2r}
  P_LIFTW1 = P_BLK0 + 24, P_LIFTW2, P_LIFTB1, P_LIFTB2, P_PW1, P_PW2, P_PB1, P_PB2,
  P_GIN_W0, P_GIN_W1, P_GIN_W2, P_GIN_B0, P_GIN_B1, P_GIN_B2,
  P_GOUT_W0, P_GOUT_W1, P_GOUT_W2, P_GOUT_B0, P_GOUT_B1, P_GOUT_B2,
  P_MPI_E = 55,   // W0,W1,W2,B0,B1,B2,BE,G
  P_MPI_N = 63,
  P_MPO_E = 71,
  P_MPO_N = 79,
  P_NI = 87,      // W0,W1,W2,B0,B1,B2,BE,G
  P_NO = 95,      // W0,W1,W2,B0,B1,B2
  P_PROJW = 101, P_PROJB = 102,
};

extern "C" void kernel_launch(void* const* d_in, const int* in_sizes, int n_in,
                              void* d_out, int out_size, void* d_ws,
                              size_t ws_size, hipStream_t stream) {
  (void)n_in; (void)out_size; (void)ws_size;
  const int N  = in_sizes[0];
  const int E  = in_sizes[5];
  const int GG = in_sizes[6] / 2;
  const int P  = in_sizes[7];

  const int*   x_type = (const int*)d_in[0];
  const float* pos    = (const float*)d_in[1];
  const float* recent = (const float*)d_in[2];
  const int*   ei     = (const int*)d_in[3];
  const float* eattr  = (const float*)d_in[4];
  const float* ndist  = (const float*)d_in[5];
  const float* lq     = (const float*)d_in[6];
  const int*   nin_q  = (const int*)d_in[7];
  const int*   nin_p  = (const int*)d_in[8];
  const int*   nout_q = (const int*)d_in[9];
  const int*   nout_p = (const int*)d_in[10];
#define PRM(i) ((const float*)d_in[11 + (i)])
  const int* src = ei;
  const int* dst = ei + E;

  // ---- workspace carve-out (floats, 16B-padded for float2/float4 paths) ----
  float* w = (float*)d_ws;
  size_t off = 0;
  auto alloc = [&](size_t n) {
    n = (n + 3) & ~(size_t)3;
    float* p = w + off; off += n; return p;
  };
  float* nf    = alloc((size_t)N * 128);
  float* eb1   = alloc((size_t)E * 128);
  float* eb2   = alloc((size_t)E * 128);
  float* ef    = alloc((size_t)E * 128);
  float* agg   = alloc((size_t)N * 128);
  float* nb1   = alloc((size_t)N * 128);
  float* nb2   = alloc((size_t)N * 128);
  float* out3  = alloc((size_t)N * 3);
  float* p2d   = alloc((size_t)N * 2);
  float* kin   = alloc((size_t)P * 5);
  float* kv1   = alloc((size_t)P * 32);
  float* kv2   = alloc((size_t)P * 64);
  float* kv3   = alloc((size_t)P * 3);
  float* kout  = alloc((size_t)P * 128);
  float* insum = alloc((size_t)GG * 3);
  float* incnt = alloc((size_t)GG);
  float* osum  = alloc((size_t)N * 128);
  float* ocnt  = alloc((size_t)N);
  float* inp   = alloc((size_t)GG * 3);
  float* fx    = alloc(1024 * 32);
  float* fh    = alloc(1024 * 32);
  float* ft    = alloc(1024 * 32);
  float* fms   = alloc(1024 * 32);
  float* fsp   = alloc(1024 * 32);
  float* ft16  = alloc(1024 * 16);
  float* fp1   = alloc(1024 * 32);
  float* latPM = alloc(1024 * 128);
  float* latCM = alloc(1024 * 128);
  const int SZF = GRID_G * GRID_G * KXN;  // 17408
  float* Yr = alloc(SZF); float* Yi = alloc(SZF);
  float* Xr = alloc(SZF); float* Xi = alloc(SZF);
  float* Or_ = alloc(SZF); float* Oi_ = alloc(SZF);
  float* Zr = alloc(SZF); float* Zi = alloc(SZF);

  // ---- pos2d = tanh(recent_pos @ W2d + b2d) ----
  pos2d_kernel<<<cdiv(N, 256), 256, 0, stream>>>(recent, PRM(P_W2D), PRM(P_B2D), p2d, N);

  // ---- node encoder: concat(embed[x_type], pos) -> 128 (relu,relu,LN) ----
  run_gemm_cat(stream, PRM(P_EMBED), x_type, 16, pos, nullptr, 21, nullptr, 0,
               PRM(P_NI + 0), PRM(P_NI + 3), nb1, N, 128, 1);
  run_gemm(stream, nb1, 128, PRM(P_NI + 1), PRM(P_NI + 4), nullptr, nb2, N, 128, 128, 1);
  run_gemm(stream, nb2, 128, PRM(P_NI + 2), PRM(P_NI + 5), nullptr, nf, N, 128, 128, 0);
  ln_kernel<<<cdiv(N, 8), 256, 0, stream>>>(nf, PRM(P_NI + 7), PRM(P_NI + 6), nullptr, N, 128);

  // ---- edge encoder: edge_attr[E,4] -> 128 (relu,relu,LN) ----
  run_gemm(stream, eattr, 4, PRM(P_EIN_W0), PRM(P_EIN_B0), nullptr, eb1, E, 4, 128, 1);
  run_gemm(stream, eb1, 128, PRM(P_EIN_W1), PRM(P_EIN_B1), nullptr, eb2, E, 128, 128, 1);
  run_gemm(stream, eb2, 128, PRM(P_EIN_W2), PRM(P_EIN_B2), nullptr, ef, E, 128, 128, 0);
  ln_kernel<<<cdiv(E, 8), 256, 0, stream>>>(ef, PRM(P_EIN_G), PRM(P_EIN_BE), nullptr, E, 128);

  // ---- SchInteractionNetwork (residual GNS block) ----
  auto sch = [&](int eb, int nb) {
    // edge MLP: fused gather concat(nf[dst], nf[src], ef), K=384
    run_gemm_cat(stream, nf, dst, 128, nf, src, 128, ef, 128,
                 PRM(eb + 0), PRM(eb + 3), eb1, E, 128, 1);
    run_gemm(stream, eb1, 128, PRM(eb + 1), PRM(eb + 4), nullptr, eb2, E, 128, 128, 1);
    run_gemm(stream, eb2, 128, PRM(eb + 2), PRM(eb + 5), nullptr, eb1, E, 128, 128, 0);
    // LN then m *= node_dist (fused row scale)
    ln_kernel<<<cdiv(E, 8), 256, 0, stream>>>(eb1, PRM(eb + 7), PRM(eb + 6), ndist, E, 128);
    // aggregate messages into dst nodes
    hipMemsetAsync(agg, 0, (size_t)N * 128 * sizeof(float), stream);
    scatter_add_kernel<<<cdiv((long long)E * 128, 256), 256, 0, stream>>>(eb1, dst, agg, E, 128);
    // node MLP: concat(nf, agg), K=256
    run_gemm_cat(stream, nf, nullptr, 128, agg, nullptr, 128, nullptr, 0,
                 PRM(nb + 0), PRM(nb + 3), nb1, N, 128, 1);
    run_gemm(stream, nb1, 128, PRM(nb + 1), PRM(nb + 4), nullptr, nb2, N, 128, 128, 1);
    run_gemm(stream, nb2, 128, PRM(nb + 2), PRM(nb + 5), nullptr, nb1, N, 128, 128, 0);
    ln_kernel<<<cdiv(N, 8), 256, 0, stream>>>(nb1, PRM(nb + 7), PRM(nb + 6), nullptr, N, 128);
    add_inplace_kernel<<<cdiv((long long)N * 128, 256), 256, 0, stream>>>(nf, nb1, (long long)N * 128);
    add_inplace_kernel<<<cdiv((long long)E * 128, 256), 256, 0, stream>>>(ef, eb1, (long long)E * 128);
  };

  auto node_out_mlp = [&](float* target) {
    run_gemm(stream, nf, 128, PRM(P_NO + 0), PRM(P_NO + 3), nullptr, nb1, N, 128, 128, 1);
    run_gemm(stream, nb1, 128, PRM(P_NO + 1), PRM(P_NO + 4), nullptr, nb2, N, 128, 128, 1);
    run_gemm(stream, nb2, 128, PRM(P_NO + 2), PRM(P_NO + 5), nullptr, target, N, 128, 3, 0);
  };

  sch(P_MPI_E, P_MPI_N);
  node_out_mlp(out3);

  // ---- in-GNO: mean_j MLP(y_j - x_i, f_y_j) over radius pairs ----
  kin_build_kernel<<<cdiv(P, 256), 256, 0, stream>>>(nin_q, nin_p, p2d, lq, out3, kin, P);
  run_gemm(stream, kin, 5, PRM(P_GIN_W0), PRM(P_GIN_B0), nullptr, kv1, P, 5, 32, 2);
  run_gemm(stream, kv1, 32, PRM(P_GIN_W1), PRM(P_GIN_B1), nullptr, kv2, P, 32, 64, 2);
  run_gemm(stream, kv2, 64, PRM(P_GIN_W2), PRM(P_GIN_B2), nullptr, kv3, P, 64, 3, 0);
  hipMemsetAsync(insum, 0, (size_t)GG * 3 * sizeof(float), stream);
  hipMemsetAsync(incnt, 0, (size_t)GG * sizeof(float), stream);
  gno_in_accum_kernel<<<cdiv((long long)P * 3, 256), 256, 0, stream>>>(kv3, nin_q, insum, incnt, P);
  div_mean_kernel<<<cdiv((long long)GG * 3, 256), 256, 0, stream>>>(insum, incnt, inp, GG, 3);

  // ---- FNO on 32x32 grid, FC=32 channels (pixel-major [HW,C]) ----
  run_gemm(stream, inp, 3, PRM(P_LIFTW1), PRM(P_LIFTB1), nullptr, fh, 1024, 3, 32, 2);
  run_gemm(stream, fh, 32, PRM(P_LIFTW2), PRM(P_LIFTB2), nullptr, fx, 1024, 32, 32, 0);
  for (int b = 0; b < 2; ++b) {
    int pb = P_BLK0 + b * 12;  // m1W,m1b,m2W,m2b,msW,msb,skW,skb,w1i,w1r,w2i,w2r
    ewise_act_kernel<<<cdiv(1024 * 32, 256), 256, 0, stream>>>(fh, fx, 1024 * 32, 2);  // h=gelu(x)
    ewise_act_kernel<<<cdiv(1024 * 32, 256), 256, 0, stream>>>(ft, fh, 1024 * 32, 3);  // tanh(h)
    dft_row_fwd_kernel<<<cdiv(SZF, 256), 256, 0, stream>>>(ft, Yr, Yi, 32);
    dft_col_fwd_kernel<<<cdiv(SZF, 256), 256, 0, stream>>>(Yr, Yi, Xr, Xi, 32);
    spec_mul_kernel<<<cdiv(SZF, 256), 256, 0, stream>>>(Xr, Xi, PRM(pb + 9), PRM(pb + 8),
                                                        PRM(pb + 11), PRM(pb + 10), Or_, Oi_);
    dft_col_inv_kernel<<<cdiv(SZF, 256), 256, 0, stream>>>(Or_, Oi_, Zr, Zi);
    dft_row_inv_kernel<<<cdiv(32 * 1024, 256), 256, 0, stream>>>(Zr, Zi, fsp, 32);
    // x = spec(tanh(h)) + c11(h, skW, skb)
    run_gemm(stream, fh, 32, PRM(pb + 6), PRM(pb + 7), fsp, fx, 1024, 32, 32, 0);
    ewise_act_kernel<<<cdiv(1024 * 32, 256), 256, 0, stream>>>(fh, fx, 1024 * 32, 2);  // h2=gelu(x)
    run_gemm(stream, fh, 32, PRM(pb + 4), PRM(pb + 5), nullptr, fms, 1024, 32, 32, 0);
    run_gemm(stream, fh, 32, PRM(pb + 0), PRM(pb + 1), nullptr, ft16, 1024, 32, 16, 2);
    run_gemm(stream, ft16, 16, PRM(pb + 2), PRM(pb + 3), fms, fx, 1024, 16, 32, 0);
  }
  run_gemm(stream, fx, 32, PRM(P_PW1), PRM(P_PB1), nullptr, fp1, 1024, 32, 32, 2);
  run_gemm(stream, fp1, 32, PRM(P_PW2), PRM(P_PB2), nullptr, latPM, 1024, 32, 128, 0);
  // bug-faithful channel-major reshape: lat[p,c] = flat(HID,G,G)[p*128+c]
  transpose_pm_cm_kernel<<<cdiv(1024 * 128, 256), 256, 0, stream>>>(latPM, latCM, 1024, 128);

  // ---- out-GNO: mean_j MLP(y_j - x_i) * lat_j ----
  kd_build_kernel<<<cdiv(P, 256), 256, 0, stream>>>(nout_q, nout_p, p2d, lq, kin, P);
  run_gemm(stream, kin, 2, PRM(P_GOUT_W0), PRM(P_GOUT_B0), nullptr, kv1, P, 2, 32, 2);
  run_gemm(stream, kv1, 32, PRM(P_GOUT_W1), PRM(P_GOUT_B1), nullptr, kv2, P, 32, 64, 2);
  run_gemm(stream, kv2, 64, PRM(P_GOUT_W2), PRM(P_GOUT_B2), nullptr, kout, P, 64, 128, 0);
  hipMemsetAsync(osum, 0, (size_t)N * 128 * sizeof(float), stream);
  hipMemsetAsync(ocnt, 0, (size_t)N * sizeof(float), stream);
  gno_out_accum_kernel<<<cdiv((long long)P * 128, 256), 256, 0, stream>>>(
      kout, nout_q, nout_p, latCM, osum, ocnt, P);
  div_mean_kernel<<<cdiv((long long)N * 128, 256), 256, 0, stream>>>(osum, ocnt, nb1, N, 128);

  // ---- projection + second Sch block + decoder ----
  run_gemm(stream, nb1, 128, PRM(P_PROJW), PRM(P_PROJB), nullptr, nf, N, 128, 128, 0);
  sch(P_MPO_E, P_MPO_N);
  node_out_mlp((float*)d_out);
}
